// MolecularGCN_79276506349569
// MI455X (gfx1250) — matrix operand order repaired and checked by reading, not compiled, not run on hardware
//
#include <hip/hip_runtime.h>
#include <math.h>

// ---------------------------------------------------------------------------
// MI455X (gfx1250) MolecularGCN forward.
//  - GEMMs: v_wmma_f32_16x16x32_f16, wave32, one 16x16 C-tile per wave.
//    A and W are pre-converted to f16 staging buffers (K zero-padded to a
//    multiple of 32, W transposed to [N,Kpad]) so the inner loop is exactly
//    4x global_load_b128 + 1x v_wmma with no divergence/guards.
//  - Edge segment ops (HBM-bound at 23.3 TB/s) use f32 global atomics.
// ---------------------------------------------------------------------------

typedef __attribute__((ext_vector_type(16))) _Float16 v16h;
typedef __attribute__((ext_vector_type(8)))  _Float16 h8;
typedef __attribute__((ext_vector_type(8)))  float    v8f;

#define TPB 256
#define NHEADS 4
#define BN_EPS 1e-5f

#define GSL(i, total) for (long long i = (long long)blockIdx.x * blockDim.x + threadIdx.x; \
                           i < (total); i += (long long)gridDim.x * blockDim.x)

__device__ __forceinline__ float leakyr(float x) { return x > 0.f ? x : 0.2f * x; }

__device__ __forceinline__ void atomicMaxFloat(float* addr, float val) {
  if (val >= 0.f) atomicMax((int*)addr, __float_as_int(val));
  else            atomicMin((unsigned int*)addr, __float_as_uint(val));
}

// ------------------------- f16 staging conversions -------------------------
// A[M,K] f32 (optionally row-scaled) -> Ah[M,Kpad] f16, zero padded.
__global__ void k_castA(const float* __restrict__ A, _Float16* __restrict__ Ah,
                        const float* __restrict__ rowScale, int M, int K, int Kpad) {
  GSL(i, (long long)M * Kpad) {
    int r = (int)(i / Kpad), k = (int)(i % Kpad);
    float v = (k < K) ? A[(size_t)r * K + k] : 0.f;
    if (rowScale) v *= rowScale[r];
    Ah[i] = (_Float16)v;
  }
}
// W[K,N] f32 -> Wt[N,Kpad] f16 transposed, zero padded.
__global__ void k_castW(const float* __restrict__ W, _Float16* __restrict__ Wt,
                        int K, int Kpad, int N) {
  GSL(i, (long long)N * Kpad) {
    int c = (int)(i / Kpad), k = (int)(i % Kpad);
    Wt[i] = (_Float16)((k < K) ? W[(size_t)k * N + c] : 0.f);
  }
}

// ------------------------------- GEMM (WMMA) -------------------------------
// C[M,N] (+)= Ah[M,Kpad] @ Wt[N,Kpad]^T + bias[N]?   (M%16==0, N%16==0)
__global__ void k_gemm_wmma(const h8* __restrict__ Ah, const h8* __restrict__ Wt,
                            const float* __restrict__ bias, float* __restrict__ C,
                            int M, int Kp8, int N, int addC) {
  int gw = (int)((blockIdx.x * (long long)blockDim.x + threadIdx.x) >> 5);
  int tilesN = N >> 4;
  if (gw >= (M >> 4) * tilesN) return;        // wave-uniform exit (EXEC all-1)
  int lane = threadIdx.x & 31;
  int half = lane >> 4;
  int l15  = lane & 15;
  int tM = gw / tilesN, tN = gw - tM * tilesN;
  const h8* arow = Ah + (size_t)((tM << 4) + l15) * Kp8;   // A row (both halves)
  const h8* brow = Wt + (size_t)((tN << 4) + l15) * Kp8;   // B column
  v8f acc = {0.f, 0.f, 0.f, 0.f, 0.f, 0.f, 0.f, 0.f};
  for (int k8 = 0; k8 < Kp8; k8 += 4) {       // 32 K-values per iteration
    h8 a0 = arow[k8 + half];                  // K = k0 + 8*half  .. +7
    h8 a1 = arow[k8 + 2 + half];              // K = k0+16+8*half .. +7
    h8 b0 = brow[k8 + half];                  // K = k0 + 8*half  .. +7
    h8 b1 = brow[k8 + 2 + half];              // K = k0+16+8*half .. +7
    v16h a = __builtin_shufflevector(a0, a1, 0,1,2,3,4,5,6,7,8,9,10,11,12,13,14,15);
    v16h b = __builtin_shufflevector(b0, b1, 0,1,2,3,4,5,6,7,8,9,10,11,12,13,14,15);
    acc = __builtin_amdgcn_wmma_f32_16x16x32_f16(false, a, false, b, (short)0, acc,
                                                 false, false);
  }
  int col = (tN << 4) + l15;
#pragma unroll
  for (int r = 0; r < 8; ++r) {
    int rr = (tM << 4) + (half << 3) + r;     // C layout: vgpr r -> row r + 8*half
    size_t o = (size_t)rr * N + col;
    float v = acc[r];
    if (bias) v += bias[col];
    if (addC) v += C[o];
    C[o] = v;
  }
}

// --------------------------- elementwise helpers ---------------------------
__global__ void k_zero(float* p, long long cnt) { GSL(i, cnt) p[i] = 0.f; }
__global__ void k_fill(float* p, float v, long long cnt) { GSL(i, cnt) p[i] = v; }
__global__ void k_add(const float* a, const float* b, float* o, long long cnt) {
  GSL(i, cnt) o[i] = a[i] + b[i];
}
__global__ void k_tanh_ip(float* p, long long cnt) { GSL(i, cnt) p[i] = tanhf(p[i]); }

// ------------------------------- degrees -----------------------------------
__global__ void k_indeg(const int* __restrict__ ei, float* indeg, int E) {
  GSL(e, (long long)E) atomicAdd(&indeg[ei[E + e]], 1.f);
}
__global__ void k_deg_final(const float* indeg, float* dinv, float* invcnt, int n) {
  GSL(v, (long long)n) {
    dinv[v]   = rsqrtf(indeg[v] + 1.f);
    invcnt[v] = 1.f / fmaxf(indeg[v], 1.f);
  }
}

// --------------------------------- GCN -------------------------------------
__global__ void k_gcn_scatter(const float* __restrict__ h, const int* __restrict__ ei,
                              const float* __restrict__ dinv, float* acc, int E, int F) {
  GSL(i, (long long)E * F) {
    int e = (int)(i / F), f = (int)(i % F);
    int s = ei[e], d = ei[E + e];
    atomicAdd(&acc[(size_t)d * F + f], dinv[s] * dinv[d] * h[(size_t)s * F + f]);
  }
}
__global__ void k_gcn_self(const float* __restrict__ h, const float* __restrict__ dinv,
                           const float* __restrict__ bias, float* acc, int n, int F) {
  GSL(i, (long long)n * F) {
    int v = (int)(i / F), f = (int)(i % F);
    acc[i] += dinv[v] * dinv[v] * h[i] + bias[f];
  }
}

// --------------------------------- GAT -------------------------------------
__global__ void k_gat_coeff(const float* __restrict__ hg, const float* __restrict__ asrc,
                            const float* __restrict__ adst, float* a_s, float* a_d,
                            int n, int F, int dh) {
  GSL(i, (long long)n * NHEADS) {
    int v = (int)(i / NHEADS), h = (int)(i % NHEADS);
    const float* hv = hg + (size_t)v * F + h * dh;
    float ss = 0.f, dd = 0.f;
    for (int d = 0; d < dh; ++d) { float x = hv[d]; ss += x * asrc[h * dh + d]; dd += x * adst[h * dh + d]; }
    a_s[i] = ss; a_d[i] = dd;
  }
}
__global__ void k_gat_init_m(const float* a_s, const float* a_d, float* m, int n) {
  GSL(i, (long long)n * NHEADS) m[i] = leakyr(a_s[i] + a_d[i]);   // self-loop logit
}
__global__ void k_gat_edge_max(const int* __restrict__ ei, const float* a_s,
                               const float* a_d, float* m, int E) {
  GSL(i, (long long)E * NHEADS) {
    int e = (int)(i >> 2), h = (int)(i & 3);
    int s = ei[e], d = ei[E + e];
    atomicMaxFloat(&m[d * NHEADS + h], leakyr(a_s[s * NHEADS + h] + a_d[d * NHEADS + h]));
  }
}
__global__ void k_gat_expsum_edges(const int* __restrict__ ei, const float* a_s,
                                   const float* a_d, const float* m, float* sden, int E) {
  GSL(i, (long long)E * NHEADS) {
    int e = (int)(i >> 2), h = (int)(i & 3);
    int s = ei[e], d = ei[E + e];
    float lg = leakyr(a_s[s * NHEADS + h] + a_d[d * NHEADS + h]);
    atomicAdd(&sden[d * NHEADS + h], expf(lg - m[d * NHEADS + h]));
  }
}
__global__ void k_gat_expsum_self(const float* a_s, const float* a_d, const float* m,
                                  float* sden, int n) {
  GSL(i, (long long)n * NHEADS) sden[i] += expf(leakyr(a_s[i] + a_d[i]) - m[i]);
}
__global__ void k_gat_agg_edges(const int* __restrict__ ei, const float* __restrict__ hg,
                                const float* a_s, const float* a_d, const float* m,
                                const float* sden, float* acc, int E, int F, int dh) {
  GSL(i, (long long)E * F) {
    int e = (int)(i / F), f = (int)(i % F);
    int s = ei[e], d = ei[E + e];
    int h = f / dh;
    float lg = leakyr(a_s[s * NHEADS + h] + a_d[d * NHEADS + h]);
    float al = expf(lg - m[d * NHEADS + h]) / (sden[d * NHEADS + h] + 1e-16f);
    atomicAdd(&acc[(size_t)d * F + f], al * hg[(size_t)s * F + f]);
  }
}
__global__ void k_gat_agg_self(const float* __restrict__ hg, const float* a_s,
                               const float* a_d, const float* m, const float* sden,
                               float* acc, int n, int F, int dh) {
  GSL(i, (long long)n * F) {
    int v = (int)(i / F), f = (int)(i % F);
    int h = f / dh;
    float lg = leakyr(a_s[v * NHEADS + h] + a_d[v * NHEADS + h]);
    float al = expf(lg - m[v * NHEADS + h]) / (sden[v * NHEADS + h] + 1e-16f);
    acc[i] += al * hg[i];
  }
}

// -------------------------------- SAGE/GIN ---------------------------------
__global__ void k_sage_scatter(const float* __restrict__ x, const int* __restrict__ ei,
                               float* sum_agg, int E, int F) {
  GSL(i, (long long)E * F) {
    int e = (int)(i / F), f = (int)(i % F);
    atomicAdd(&sum_agg[(size_t)ei[E + e] * F + f], x[(size_t)ei[e] * F + f]);
  }
}
__global__ void k_bn_relu(float* z, const float* g, const float* b, const float* rm,
                          const float* rv, long long rows, int F) {
  GSL(i, rows * F) {
    int f = (int)(i % F);
    float v = (z[i] - rm[f]) * g[f] * rsqrtf(rv[f] + BN_EPS) + b[f];
    z[i] = fmaxf(v, 0.f);
  }
}

// ------------------------------ layer epilogue ------------------------------
__global__ void k_layer_final(const float* __restrict__ acc, const float* __restrict__ bgat,
                              const float* g, const float* b, const float* rm,
                              const float* rv, float* xout, int n, int F) {
  GSL(i, (long long)n * F) {
    int f = (int)(i % F);
    float v = (acc[i] + bgat[f]) * 0.25f;
    v = (v - rm[f]) * g[f] * rsqrtf(rv[f] + BN_EPS) + b[f];
    xout[i] = fmaxf(v, 0.f);
  }
}

// ------------------------------- pooling -----------------------------------
__global__ void k_att_score(const float* __restrict__ t, const float* __restrict__ W2,
                            const float* __restrict__ b2, float* sc, int n, int K) {
  GSL(v, (long long)n) {
    float s = 0.f;
    const float* tv = t + (size_t)v * K;
    for (int k = 0; k < K; ++k) s += tv[k] * W2[k];
    sc[v] = s + b2[0];
  }
}
__global__ void k_amax(const float* sc, float* gmax, int n) {
  GSL(v, (long long)n) atomicMaxFloat(gmax, sc[v]);
}
__global__ void k_expsum(const float* sc, const float* gmax, float* aw, float* gsum, int n) {
  GSL(v, (long long)n) { float e = expf(sc[v] - gmax[0]); aw[v] = e; atomicAdd(gsum, e); }
}
__global__ void k_awnorm(float* aw, const float* gsum, const int* batch, float* segw, int n) {
  GSL(v, (long long)n) { float a = aw[v] / gsum[0]; aw[v] = a; atomicAdd(&segw[batch[v]], a); }
}
__global__ void k_cnt(const int* batch, float* cnt, int n) {
  GSL(v, (long long)n) atomicAdd(&cnt[batch[v]], 1.f);
}
__global__ void k_comb_init(float* comb, int B) {
  GSL(i, (long long)B * 2048) {
    int f = (int)(i % 2048);
    comb[i] = (f >= 1024 && f < 1536) ? -__builtin_inff() : 0.f;  // max-pool section
  }
}
__global__ void k_pool_scatter(const float* __restrict__ x, const float* aw,
                               const float* segw, const int* batch, float* comb,
                               int n, int F) {
  GSL(i, (long long)n * F) {
    int v = (int)(i / F), f = (int)(i % F);
    int g = batch[v];
    float xv = x[i];
    float w = aw[v] / (segw[g] + 1e-8f);
    float* base = comb + (size_t)g * 2048;
    atomicAdd(&base[f], w * xv);              // graph_emb  [0..511]
    atomicMaxFloat(&base[1024 + f], xv);      // max_pool   [1024..1535]
    atomicAdd(&base[1536 + f], xv);           // add_pool   [1536..2047]
  }
}
__global__ void k_comb_final(float* comb, const float* cnt, int B) {
  GSL(i, (long long)B * 512) {
    int g = (int)(i / 512), f = (int)(i % 512);
    float* base = comb + (size_t)g * 2048;
    base[512 + f] = base[1536 + f] / fmaxf(cnt[g], 1.f);   // mean_pool
    float mx = base[1024 + f];
    if (!(mx > -__builtin_inff())) base[1024 + f] = 0.f;   // empty graph -> 0
  }
}

// --------------------------------- head ------------------------------------
__global__ void k_bn_gelu(float* z, const float* g, const float* b, const float* rm,
                          const float* rv, long long rows, int F) {
  GSL(i, rows * F) {
    int f = (int)(i % F);
    float v = (z[i] - rm[f]) * g[f] * rsqrtf(rv[f] + BN_EPS) + b[f];
    z[i] = 0.5f * v * (1.f + erff(v * 0.70710678118654752f));  // exact GELU
  }
}
__global__ void k_head_out(const float* __restrict__ h3, const float* __restrict__ W4,
                           const float* __restrict__ b4, float* out, int B, int K) {
  GSL(g, (long long)B) {
    float s = 0.f;
    const float* hv = h3 + (size_t)g * K;
    for (int k = 0; k < K; ++k) s += hv[k] * W4[k];
    out[g] = s + b4[0];
  }
}

// ------------------------------- host side ----------------------------------
static inline dim3 gsGrid(long long total) {
  long long bl = (total + TPB - 1) / TPB;
  if (bl > 1048576) bl = 1048576;
  if (bl < 1) bl = 1;
  return dim3((unsigned)bl);
}

struct GemmCtx { _Float16* Ah; _Float16* Wt; hipStream_t st; };

// C[M,N] (+)= (A * rowScale?) @ W + bias?   via f16 staging + WMMA
static inline void gemm(const GemmCtx& g, const float* A, const float* W,
                        const float* bias, const float* rowScale, float* C,
                        int M, int K, int N, int addC) {
  int Kpad = (K + 31) & ~31;
  k_castA<<<gsGrid((long long)M * Kpad), TPB, 0, g.st>>>(A, g.Ah, rowScale, M, K, Kpad);
  k_castW<<<gsGrid((long long)N * Kpad), TPB, 0, g.st>>>(W, g.Wt, K, Kpad, N);
  long long waves = (long long)(M / 16) * (N / 16);
  long long thr = waves * 32;
  k_gemm_wmma<<<dim3((unsigned)((thr + TPB - 1) / TPB)), dim3(TPB), 0, g.st>>>(
      (const h8*)g.Ah, (const h8*)g.Wt, bias, C, M, Kpad / 8, N, addC);
}

extern "C" void kernel_launch(void* const* d_in, const int* in_sizes, int n_in,
                              void* d_out, int out_size, void* d_ws, size_t ws_size,
                              hipStream_t stream) {
  (void)in_sizes; (void)n_in; (void)out_size; (void)ws_size;
  const int n = 20000, E = 320000, B = 512;

  // ---- inputs (setup_inputs dict order, params flattened per-key insertion order)
  int p = 0;
  const float* X     = (const float*)d_in[p++];   // [n,36]
  const int*   EI    = (const int*)  d_in[p++];   // [2,E]
  const int*   BATCH = (const int*)  d_in[p++];   // [n]
  const float *gcnW[3], *gcnb[3];
  for (int i = 0; i < 3; ++i) { gcnW[i] = (const float*)d_in[p++]; gcnb[i] = (const float*)d_in[p++]; }
  const float *gatW[3], *gatb[3], *gatAS[3], *gatAD[3];
  for (int i = 0; i < 3; ++i) {
    gatW[i]  = (const float*)d_in[p++]; gatb[i]  = (const float*)d_in[p++];
    gatAS[i] = (const float*)d_in[p++]; gatAD[i] = (const float*)d_in[p++];
  }
  const float *sageLW[3], *sageLb[3], *sageRW[3], *sageRb[3];
  for (int i = 0; i < 3; ++i) {
    sageLW[i] = (const float*)d_in[p++]; sageLb[i] = (const float*)d_in[p++];
    sageRW[i] = (const float*)d_in[p++]; sageRb[i] = (const float*)d_in[p++];
  }
  const float *ginW1[3], *ginb1[3], *ginBNg[3], *ginBNb[3], *ginBNrm[3], *ginBNrv[3], *ginW2[3], *ginb2[3];
  for (int i = 0; i < 3; ++i) {
    ginW1[i]  = (const float*)d_in[p++]; ginb1[i]  = (const float*)d_in[p++];
    ginBNg[i] = (const float*)d_in[p++]; ginBNb[i] = (const float*)d_in[p++];
    ginBNrm[i]= (const float*)d_in[p++]; ginBNrv[i]= (const float*)d_in[p++];
    ginW2[i]  = (const float*)d_in[p++]; ginb2[i]  = (const float*)d_in[p++];
  }
  const float *bnG[3], *bnB[3], *bnRM[3], *bnRV[3];
  for (int i = 0; i < 3; ++i) {
    bnG[i] = (const float*)d_in[p++]; bnB[i] = (const float*)d_in[p++];
    bnRM[i]= (const float*)d_in[p++]; bnRV[i]= (const float*)d_in[p++];
  }
  const float* attW1 = (const float*)d_in[p++]; const float* attb1 = (const float*)d_in[p++];
  const float* attW2 = (const float*)d_in[p++]; const float* attb2 = (const float*)d_in[p++];
  const float* hW1 = (const float*)d_in[p++]; const float* hb1 = (const float*)d_in[p++];
  const float* h1g = (const float*)d_in[p++]; const float* h1b = (const float*)d_in[p++];
  const float* h1rm= (const float*)d_in[p++]; const float* h1rv= (const float*)d_in[p++];
  const float* hW2 = (const float*)d_in[p++]; const float* hb2 = (const float*)d_in[p++];
  const float* h2g = (const float*)d_in[p++]; const float* h2b = (const float*)d_in[p++];
  const float* h2rm= (const float*)d_in[p++]; const float* h2rv= (const float*)d_in[p++];
  const float* hW3 = (const float*)d_in[p++]; const float* hb3 = (const float*)d_in[p++];
  const float* h3g = (const float*)d_in[p++]; const float* h3b = (const float*)d_in[p++];
  const float* h3rm= (const float*)d_in[p++]; const float* h3rv= (const float*)d_in[p++];
  const float* hW4 = (const float*)d_in[p++]; const float* hb4 = (const float*)d_in[p++];

  // ---- workspace (bump allocator; ~300 MB)
  float* wp = (float*)d_ws;
  float* indeg  = wp; wp += n;
  float* dinv   = wp; wp += n;
  float* invcnt = wp; wp += n;
  float* bufA = wp; wp += (size_t)n * 512;    // current x
  float* bufB = wp; wp += (size_t)n * 512;    // conv-sum accumulator
  float* bufC = wp; wp += (size_t)n * 512;    // h / hg / sum_agg
  float* bufD = wp; wp += (size_t)n * 512;    // gin_in / att hidden
  float* bufE = wp; wp += (size_t)n * 1024;   // GIN hidden
  float* a_s  = wp; wp += (size_t)n * NHEADS;
  float* a_d  = wp; wp += (size_t)n * NHEADS;
  float* mb   = wp; wp += (size_t)n * NHEADS;
  float* sden = wp; wp += (size_t)n * NHEADS;
  float* sc   = wp; wp += n;
  float* aw   = wp; wp += n;
  float* gmax = wp; wp += 1;
  float* gsum = wp; wp += 1;
  float* segw = wp; wp += B;
  float* cnt  = wp; wp += B;
  float* comb = wp; wp += (size_t)B * 2048;
  float* hh1  = wp; wp += (size_t)B * 512;
  float* hh2  = wp; wp += (size_t)B * 256;
  float* hh3  = wp; wp += (size_t)B * 128;
  wp += (16 - ((wp - (float*)d_ws) & 15)) & 15;          // 64B align staging
  _Float16* Ah = (_Float16*)wp; wp += (size_t)n * 1024 / 2;   // f16 A staging
  _Float16* Wt = (_Float16*)wp; wp += (size_t)2048 * 512 / 2; // f16 W^T staging
  float* outF = (float*)d_out;
  GemmCtx G{Ah, Wt, stream};

  // ---- degrees / norms
  k_zero<<<gsGrid(n), TPB, 0, stream>>>(indeg, n);
  k_indeg<<<gsGrid(E), TPB, 0, stream>>>(EI, indeg, E);
  k_deg_final<<<gsGrid(n), TPB, 0, stream>>>(indeg, dinv, invcnt, n);

  const int fiA[3] = {36, 128, 256};
  const int foA[3] = {128, 256, 512};
  const float* xin = X;
  for (int L = 0; L < 3; ++L) {
    int fi = fiA[L], fo = foA[L], dh = fo / NHEADS;
    long long nf = (long long)n * fo;
    k_zero<<<gsGrid(nf), TPB, 0, stream>>>(bufB, nf);
    // GCN: h = xW ; acc += segsum(norm_e * h[src]) + dinv^2*h + b
    gemm(G, xin, gcnW[L], nullptr, nullptr, bufC, n, fi, fo, 0);
    k_gcn_scatter<<<gsGrid((long long)E * fo), TPB, 0, stream>>>(bufC, EI, dinv, bufB, E, fo);
    k_gcn_self<<<gsGrid(nf), TPB, 0, stream>>>(bufC, dinv, gcnb[L], bufB, n, fo);
    // GAT: hg = xW ; segment softmax over dst (with self loops) ; acc += agg
    gemm(G, xin, gatW[L], nullptr, nullptr, bufC, n, fi, fo, 0);
    k_gat_coeff<<<gsGrid((long long)n * NHEADS), TPB, 0, stream>>>(bufC, gatAS[L], gatAD[L], a_s, a_d, n, fo, dh);
    k_gat_init_m<<<gsGrid((long long)n * NHEADS), TPB, 0, stream>>>(a_s, a_d, mb, n);
    k_gat_edge_max<<<gsGrid((long long)E * NHEADS), TPB, 0, stream>>>(EI, a_s, a_d, mb, E);
    k_zero<<<gsGrid((long long)n * NHEADS), TPB, 0, stream>>>(sden, (long long)n * NHEADS);
    k_gat_expsum_edges<<<gsGrid((long long)E * NHEADS), TPB, 0, stream>>>(EI, a_s, a_d, mb, sden, E);
    k_gat_expsum_self<<<gsGrid((long long)n * NHEADS), TPB, 0, stream>>>(a_s, a_d, mb, sden, n);
    k_gat_agg_edges<<<gsGrid((long long)E * fo), TPB, 0, stream>>>(EI, bufC, a_s, a_d, mb, sden, bufB, E, fo, dh);
    k_gat_agg_self<<<gsGrid(nf), TPB, 0, stream>>>(bufC, a_s, a_d, mb, sden, bufB, n, fo, dh);
    // SAGE: acc += (sum_agg/cnt)Wl + bl + xWr
    long long nfi = (long long)n * fi;
    k_zero<<<gsGrid(nfi), TPB, 0, stream>>>(bufC, nfi);
    k_sage_scatter<<<gsGrid((long long)E * fi), TPB, 0, stream>>>(xin, EI, bufC, E, fi);
    gemm(G, bufC, sageLW[L], sageLb[L], invcnt, bufB, n, fi, fo, 1);
    gemm(G, xin, sageRW[L], nullptr, nullptr, bufB, n, fi, fo, 1);
    (void)sageRb;
    // GIN: acc += MLP(sum_agg + x)
    k_add<<<gsGrid(nfi), TPB, 0, stream>>>(bufC, xin, bufD, nfi);
    gemm(G, bufD, ginW1[L], ginb1[L], nullptr, bufE, n, fi, 2 * fo, 0);
    k_bn_relu<<<gsGrid((long long)n * 2 * fo), TPB, 0, stream>>>(bufE, ginBNg[L], ginBNb[L], ginBNrm[L], ginBNrv[L], n, 2 * fo);
    gemm(G, bufE, ginW2[L], ginb2[L], nullptr, bufB, n, 2 * fo, fo, 1);
    // combine: x = relu(bn((acc + b_gat)/4))
    k_layer_final<<<gsGrid(nf), TPB, 0, stream>>>(bufB, gatb[L], bnG[L], bnB[L], bnRM[L], bnRV[L], bufA, n, fo);
    xin = bufA;
  }

  // ---- attention pooling (global softmax, per-graph renorm) + 3 pools
  gemm(G, bufA, attW1, attb1, nullptr, bufD, n, 512, 256, 0);
  k_tanh_ip<<<gsGrid((long long)n * 256), TPB, 0, stream>>>(bufD, (long long)n * 256);
  k_att_score<<<gsGrid(n), TPB, 0, stream>>>(bufD, attW2, attb2, sc, n, 256);
  k_fill<<<dim3(1), dim3(1), 0, stream>>>(gmax, -__builtin_inff(), 1);
  k_fill<<<dim3(1), dim3(1), 0, stream>>>(gsum, 0.f, 1);
  k_amax<<<gsGrid(n), TPB, 0, stream>>>(sc, gmax, n);
  k_expsum<<<gsGrid(n), TPB, 0, stream>>>(sc, gmax, aw, gsum, n);
  k_zero<<<gsGrid(B), TPB, 0, stream>>>(segw, B);
  k_zero<<<gsGrid(B), TPB, 0, stream>>>(cnt, B);
  k_awnorm<<<gsGrid(n), TPB, 0, stream>>>(aw, gsum, BATCH, segw, n);
  k_cnt<<<gsGrid(n), TPB, 0, stream>>>(BATCH, cnt, n);
  k_comb_init<<<gsGrid((long long)B * 2048), TPB, 0, stream>>>(comb, B);
  k_pool_scatter<<<gsGrid((long long)n * 512), TPB, 0, stream>>>(bufA, aw, segw, BATCH, comb, n, 512);
  k_comb_final<<<gsGrid((long long)B * 512), TPB, 0, stream>>>(comb, cnt, B);

  // ---- prediction head
  gemm(G, comb, hW1, hb1, nullptr, hh1, B, 2048, 512, 0);
  k_bn_gelu<<<gsGrid((long long)B * 512), TPB, 0, stream>>>(hh1, h1g, h1b, h1rm, h1rv, B, 512);
  gemm(G, hh1, hW2, hb2, nullptr, hh2, B, 512, 256, 0);
  k_bn_gelu<<<gsGrid((long long)B * 256), TPB, 0, stream>>>(hh2, h2g, h2b, h2rm, h2rv, B, 256);
  gemm(G, hh2, hW3, hb3, nullptr, hh3, B, 256, 128, 0);
  k_bn_gelu<<<gsGrid((long long)B * 128), TPB, 0, stream>>>(hh3, h3g, h3b, h3rm, h3rv, B, 128);
  k_head_out<<<gsGrid(B), TPB, 0, stream>>>(hh3, hW4, hb4, outF, B, 128);
}
